// MoleculeGNN_18442589569774
// MI455X (gfx1250) — compile-verified, matrix operand order checked by reading, not run on hardware
//
#include <hip/hip_runtime.h>
#include <hip/hip_bf16.h>

typedef __attribute__((ext_vector_type(16))) _Float16 v16h;
typedef __attribute__((ext_vector_type(8)))  float    v8f;

#define NCH 128   // hidden channels
#define EPSV 1e-5f

// ---------------------------------------------------------------------------
// Small utility kernels
// ---------------------------------------------------------------------------
__global__ void zero_f32_kernel(float* p, int n) {
    int i = blockIdx.x * blockDim.x + threadIdx.x;
    if (i < n) p[i] = 0.0f;
}

// x [N,9] f32 -> xf16 [N,32] f16, zero-padded cols 9..31
__global__ void conv_x_kernel(const float* __restrict__ x, _Float16* __restrict__ xf, int n32) {
    int i = blockIdx.x * blockDim.x + threadIdx.x;
    if (i >= n32) return;
    int n = i >> 5, c = i & 31;
    xf[i] = (c < 9) ? (_Float16)x[n * 9 + c] : (_Float16)0.0f;
}

// w0 [9,128] f32 -> [32,128] f16 zero-padded rows
__global__ void conv_w0_kernel(const float* __restrict__ w, _Float16* __restrict__ wf) {
    int i = blockIdx.x * blockDim.x + threadIdx.x;
    if (i >= 32 * NCH) return;
    int k = i / NCH;
    wf[i] = (k < 9) ? (_Float16)w[i] : (_Float16)0.0f;
}

__global__ void conv_f32_to_f16_kernel(const float* __restrict__ s, _Float16* __restrict__ d, int n) {
    int i = blockIdx.x * blockDim.x + threadIdx.x;
    if (i < n) d[i] = (_Float16)s[i];
}

// ---------------------------------------------------------------------------
// Degree / normalization:   deg[i] = 1 (self loop) + in-degree;  dinv = rsqrt(max(deg,1))
// ---------------------------------------------------------------------------
__global__ void deg_init_kernel(float* deg, int N) {
    int i = blockIdx.x * blockDim.x + threadIdx.x;
    if (i < N) deg[i] = 1.0f;
}
__global__ void deg_acc_kernel(const long long* __restrict__ ei, float* deg, int E) {
    int e = blockIdx.x * blockDim.x + threadIdx.x;
    if (e < E) atomicAdd(&deg[(int)ei[(size_t)E + e]], 1.0f);
}
__global__ void deg_fin_kernel(float* deg, int N) {
    int i = blockIdx.x * blockDim.x + threadIdx.x;
    if (i < N) deg[i] = rsqrtf(fmaxf(deg[i], 1.0f));
}

// ---------------------------------------------------------------------------
// WMMA GEMM: C[N,128] (f32) = A[N,K] (f16) x W[K,128] (f16).  K in {32,128}.
// One wave -> 16x128 tile.  W staged transposed in LDS: ldsW[n*K + k].
// ---------------------------------------------------------------------------
__global__ __launch_bounds__(256) void gemm_wmma_kernel(
        const _Float16* __restrict__ A, const _Float16* __restrict__ W,
        float* __restrict__ C, int nRows, int K) {
    __shared__ _Float16 ldsW[NCH * NCH];   // 32 KB max
    const int tid = threadIdx.x;
    const int total = K * NCH;
    for (int i = tid; i < total; i += 256) {
        int n = i / K, k = i - n * K;                  // transposed layout [n][k]
        ldsW[n * K + k] = W[k * NCH + n];
    }
    __syncthreads();

    const int wave = tid >> 5;
    const int lane = tid & 31;
    const int tile = blockIdx.x * 8 + wave;
    const int row0 = tile * 16;
    if (row0 >= nRows) return;                         // wave-uniform: EXEC stays all-ones

    const int m   = lane & 15;                         // A row / B col within tile
    const int grp = lane >> 4;                         // k-group select per ISA layout

    v8f acc[8];
    v8f zero = {};
#pragma unroll
    for (int t = 0; t < 8; ++t) acc[t] = zero;

    int rr = row0 + m; if (rr >= nRows) rr = nRows - 1;
    const _Float16* aRow = A + (size_t)rr * K;
    const int kts = K >> 5;

    for (int kt = 0; kt < kts; ++kt) {
        v16h a;
#pragma unroll
        for (int v = 0; v < 8; ++v) {
            int kp = kt * 32 + 16 * (v >> 2) + 8 * grp + 2 * (v & 3);
            a[2 * v]     = aRow[kp];
            a[2 * v + 1] = aRow[kp + 1];
        }
#pragma unroll
        for (int nt = 0; nt < 8; ++nt) {
            const _Float16* bCol = ldsW + (size_t)(nt * 16 + m) * K;
            v16h b;
#pragma unroll
            for (int v = 0; v < 8; ++v) {
                int kp = kt * 32 + 16 * (v >> 2) + 8 * grp + 2 * (v & 3);
                b[2 * v]     = bCol[kp];
                b[2 * v + 1] = bCol[kp + 1];
            }
            acc[nt] = __builtin_amdgcn_wmma_f32_16x16x32_f16(
                false, a, false, b, (short)0, acc[nt], false, false);
        }
    }

    // C layout: lanes 0-15 -> M = vgpr (0..7), lanes 16-31 -> M = vgpr+8
    const int rbase = row0 + grp * 8;
#pragma unroll
    for (int nt = 0; nt < 8; ++nt) {
        int col = nt * 16 + m;
#pragma unroll
        for (int v = 0; v < 8; ++v) {
            int r = rbase + v;
            if (r < nRows) C[(size_t)r * NCH + col] = acc[nt][v];
        }
    }
}

// ---------------------------------------------------------------------------
// agg = hw * dinv^2 (self loop) + bias
// ---------------------------------------------------------------------------
__global__ void self_bias_kernel(const float* __restrict__ hw, const float* __restrict__ dinv,
                                 const float* __restrict__ bias, float* __restrict__ agg, int N) {
    int t = blockIdx.x * blockDim.x + threadIdx.x;
    if (t >= N * 32) return;
    int n = t >> 5, c = (t & 31) * 4;
    float s = dinv[n]; s *= s;
    const float4 v = *(const float4*)(hw + (size_t)n * NCH + c);
    const float4 b = *(const float4*)(bias + c);
    float4 o;
    o.x = v.x * s + b.x; o.y = v.y * s + b.y; o.z = v.z * s + b.z; o.w = v.w * s + b.w;
    *(float4*)(agg + (size_t)n * NCH + c) = o;
}

// agg[dst] += hw[src] * dinv[src]*dinv[dst]   (32 threads per edge, 4 ch each)
__global__ void edge_scatter_kernel(const long long* __restrict__ ei,
                                    const float* __restrict__ hw,
                                    const float* __restrict__ dinv,
                                    float* __restrict__ agg, int E) {
    int t = blockIdx.x * blockDim.x + threadIdx.x;
    int e = t >> 5;
    if (e >= E) return;
    int c = (t & 31) * 4;
    int s = (int)ei[e];
    int d = (int)ei[(size_t)E + e];
    float w = dinv[s] * dinv[d];
    const float4 v = *(const float4*)(hw + (size_t)s * NCH + c);
    float* p = agg + (size_t)d * NCH + c;
    atomicAdd(p + 0, v.x * w);
    atomicAdd(p + 1, v.y * w);
    atomicAdd(p + 2, v.z * w);
    atomicAdd(p + 3, v.w * w);
}

// ---------------------------------------------------------------------------
// BatchNorm (training forward, biased var) over N rows, 128 channels
// ---------------------------------------------------------------------------
__global__ void bn_stats_kernel(const float* __restrict__ agg, float* __restrict__ sums, int N) {
    int c = threadIdx.x;                    // 128 threads
    int r0 = blockIdx.x * 1024;
    int r1 = r0 + 1024; if (r1 > N) r1 = N;
    float s = 0.0f, s2 = 0.0f;
    for (int r = r0; r < r1; ++r) {
        float v = agg[(size_t)r * NCH + c];
        s += v; s2 += v * v;
    }
    atomicAdd(&sums[c], s);
    atomicAdd(&sums[NCH + c], s2);
}

__global__ void bn_finalize_kernel(const float* __restrict__ sums,
                                   const float* __restrict__ g, const float* __restrict__ beta,
                                   float* __restrict__ ss, float Nf) {
    int c = threadIdx.x;                    // 128 threads
    float mu  = sums[c] / Nf;
    float var = sums[NCH + c] / Nf - mu * mu;
    float rstd = rsqrtf(var + EPSV);
    float sc = g[c] * rstd;
    ss[c]       = sc;
    ss[NCH + c] = beta[c] - mu * sc;
}

// relu(agg*scale+shift) -> hf32 (pooling) and hf16 (next GEMM)
__global__ void bn_apply_kernel(const float* __restrict__ agg, const float* __restrict__ ss,
                                float* __restrict__ hf32, _Float16* __restrict__ hf16, int total) {
    int i = blockIdx.x * blockDim.x + threadIdx.x;
    if (i >= total) return;
    int c = i & (NCH - 1);
    float v = fmaxf(agg[i] * ss[c] + ss[NCH + c], 0.0f);
    hf32[i] = v;
    hf16[i] = (_Float16)v;
}

// ---------------------------------------------------------------------------
// Pooling (batch ids sorted, values >= 0 after ReLU so uint-max == float-max)
// ---------------------------------------------------------------------------
__global__ void cnt_kernel(const long long* __restrict__ batch, float* cnt, int N) {
    int i = blockIdx.x * blockDim.x + threadIdx.x;
    if (i < N) atomicAdd(&cnt[(int)batch[i]], 1.0f);
}

__global__ void pool_kernel(const float* __restrict__ h, const long long* __restrict__ batch,
                            float* __restrict__ psum, unsigned int* __restrict__ pmax, int total) {
    int i = blockIdx.x * blockDim.x + threadIdx.x;
    if (i >= total) return;
    int n = i >> 7, c = i & (NCH - 1);
    int g = (int)batch[n];
    float v = h[i];
    atomicAdd(&psum[(size_t)g * NCH + c], v);
    atomicMax(&pmax[(size_t)g * NCH + c], __float_as_uint(v));
}

__global__ void xg_kernel(const float* __restrict__ psum, const unsigned int* __restrict__ pmax,
                          const float* __restrict__ cnt, float* __restrict__ xg, int total) {
    int i = blockIdx.x * blockDim.x + threadIdx.x;
    if (i >= total) return;
    int g = i >> 7, c = i & (NCH - 1);
    float cn = fmaxf(cnt[g], 1.0f);
    xg[(size_t)g * 256 + c]       = psum[i] / cn;
    xg[(size_t)g * 256 + NCH + c] = __uint_as_float(pmax[i]);
}

// ---------------------------------------------------------------------------
// MLP head (tiny): out[g,j] = act( sum_k in[g,k]*w[k,j] + b[j] )
// ---------------------------------------------------------------------------
__global__ void mlp_fc_kernel(const float* __restrict__ in, const float* __restrict__ w,
                              const float* __restrict__ b, float* __restrict__ out,
                              int G, int Kin, int Kout, int doRelu) {
    int i = blockIdx.x * blockDim.x + threadIdx.x;
    if (i >= G * Kout) return;
    int g = i / Kout, j = i - g * Kout;
    float s = b[j];
    const float* row = in + (size_t)g * Kin;
    for (int k = 0; k < Kin; ++k) s += row[k] * w[(size_t)k * Kout + j];
    if (doRelu) s = fmaxf(s, 0.0f);
    out[i] = s;
}

// ---------------------------------------------------------------------------
// Host launch
// ---------------------------------------------------------------------------
extern "C" void kernel_launch(void* const* d_in, const int* in_sizes, int n_in,
                              void* d_out, int out_size, void* d_ws, size_t ws_size,
                              hipStream_t stream) {
    const float*     x     = (const float*)d_in[0];
    const long long* ei    = (const long long*)d_in[1];
    const long long* batch = (const long long*)d_in[2];
    const float*     w0    = (const float*)d_in[3];
    const float*     w12   = (const float*)d_in[4];
    const float*     cb    = (const float*)d_in[5];
    const float*     bng   = (const float*)d_in[6];
    const float*     bnb   = (const float*)d_in[7];
    const float*     m0w   = (const float*)d_in[8];
    const float*     m0b   = (const float*)d_in[9];
    const float*     m1w   = (const float*)d_in[10];
    const float*     m1b   = (const float*)d_in[11];
    const float*     m2w   = (const float*)d_in[12];
    const float*     m2b   = (const float*)d_in[13];
    float*           out   = (float*)d_out;

    const int N = in_sizes[0] / 9;
    const int E = in_sizes[1] / 2;
    const int G = out_size;

    // ---- workspace layout (256B-aligned slabs) ----
    char* ws = (char*)d_ws;
    size_t off = 0;
    auto grab = [&](size_t bytes) -> char* {
        char* p = ws + off;
        off = (off + bytes + 255) & ~(size_t)255;
        return p;
    };
    float*        HW    = (float*)grab((size_t)N * NCH * 4);   // GEMM out, then relu f32 (aliased)
    float*        AGG   = (float*)grab((size_t)N * NCH * 4);
    _Float16*     HF16  = (_Float16*)grab((size_t)N * NCH * 2);
    _Float16*     XF16  = (_Float16*)grab((size_t)N * 32 * 2);
    float*        DINV  = (float*)grab((size_t)N * 4);
    _Float16*     WF0   = (_Float16*)grab((size_t)32 * NCH * 2);
    _Float16*     WF12  = (_Float16*)grab((size_t)2 * NCH * NCH * 2);
    float*        SUMS  = (float*)grab(2 * NCH * 4);
    float*        SS    = (float*)grab(2 * NCH * 4);
    float*        PSUM  = (float*)grab((size_t)G * NCH * 4);
    unsigned int* PMAX  = (unsigned int*)grab((size_t)G * NCH * 4);
    float*        CNT   = (float*)grab((size_t)G * 4);
    float*        XG    = (float*)grab((size_t)G * 256 * 4);
    float*        MH0   = (float*)grab((size_t)G * 256 * 4);
    float*        MH1   = (float*)grab((size_t)G * 64 * 4);
    (void)ws_size; (void)n_in;

    const int T = 256;
    auto cdiv = [](long long a, long long b) { return (unsigned int)((a + b - 1) / b); };

    // ---- preprocessing ----
    conv_x_kernel<<<cdiv((long long)N * 32, T), T, 0, stream>>>(x, XF16, N * 32);
    conv_w0_kernel<<<cdiv(32 * NCH, T), T, 0, stream>>>(w0, WF0);
    conv_f32_to_f16_kernel<<<cdiv(2 * NCH * NCH, T), T, 0, stream>>>(w12, WF12, 2 * NCH * NCH);
    deg_init_kernel<<<cdiv(N, T), T, 0, stream>>>(DINV, N);
    deg_acc_kernel<<<cdiv(E, T), T, 0, stream>>>(ei, DINV, E);
    deg_fin_kernel<<<cdiv(N, T), T, 0, stream>>>(DINV, N);

    // ---- 3 GCN layers ----
    const int tiles = (N + 15) / 16;
    for (int l = 0; l < 3; ++l) {
        const _Float16* Ain = (l == 0) ? XF16 : HF16;
        const int K = (l == 0) ? 32 : NCH;
        const _Float16* Wl = (l == 0) ? WF0 : (WF12 + (size_t)(l - 1) * NCH * NCH);

        gemm_wmma_kernel<<<cdiv(tiles, 8), 256, 0, stream>>>(Ain, Wl, HW, N, K);
        self_bias_kernel<<<cdiv((long long)N * 32, T), T, 0, stream>>>(HW, DINV, cb + l * NCH, AGG, N);
        edge_scatter_kernel<<<cdiv((long long)E * 32, T), T, 0, stream>>>(ei, HW, DINV, AGG, E);
        zero_f32_kernel<<<1, 256, 0, stream>>>(SUMS, 2 * NCH);
        bn_stats_kernel<<<cdiv(N, 1024), NCH, 0, stream>>>(AGG, SUMS, N);
        bn_finalize_kernel<<<1, NCH, 0, stream>>>(SUMS, bng + l * NCH, bnb + l * NCH, SS, (float)N);
        bn_apply_kernel<<<cdiv((long long)N * NCH, T), T, 0, stream>>>(AGG, SS, HW, HF16, N * NCH);
    }

    // ---- pooling ----
    zero_f32_kernel<<<cdiv((long long)G * NCH, T), T, 0, stream>>>(PSUM, G * NCH);
    zero_f32_kernel<<<cdiv((long long)G * NCH, T), T, 0, stream>>>((float*)PMAX, G * NCH);
    zero_f32_kernel<<<cdiv(G, T), T, 0, stream>>>(CNT, G);
    cnt_kernel<<<cdiv(N, T), T, 0, stream>>>(batch, CNT, N);
    pool_kernel<<<cdiv((long long)N * NCH, T), T, 0, stream>>>(HW, batch, PSUM, PMAX, N * NCH);
    xg_kernel<<<cdiv((long long)G * NCH, T), T, 0, stream>>>(PSUM, PMAX, CNT, XG, G * NCH);

    // ---- MLP head ----
    mlp_fc_kernel<<<cdiv((long long)G * 256, T), T, 0, stream>>>(XG, m0w, m0b, MH0, G, 256, 256, 1);
    mlp_fc_kernel<<<cdiv((long long)G * 64, T), T, 0, stream>>>(MH0, m1w, m1b, MH1, G, 256, 64, 1);
    mlp_fc_kernel<<<cdiv(G, T), T, 0, stream>>>(MH1, m2w, m2b, out, G, 64, 1, 0);
}
